// GatedDeltaNet_72499047957239
// MI455X (gfx1250) — matrix operand order repaired and checked by reading, not compiled, not run on hardware
//
#include <hip/hip_runtime.h>
#include <hip/hip_bf16.h>
#include <math.h>

typedef __attribute__((ext_vector_type(16))) _Float16 v16h;
typedef __attribute__((ext_vector_type(8)))  _Float16 v8h;
typedef __attribute__((ext_vector_type(8)))  float    v8f;
typedef __attribute__((ext_vector_type(4)))  float    v4f;

#define BB   2
#define TT   2048
#define HIDD 2048
#define HH   6
#define DKk  256
#define DVv  512
#define MM   (BB * TT)        // 4096 tokens
#define CQK  (HH * DKk)       // 1536
#define CV   (HH * DVv)       // 3072

// ---------------- f32 -> f16 cast ----------------
__global__ __launch_bounds__(256)
void gdn_cast16(const float* __restrict__ in, _Float16* __restrict__ out, size_t n) {
    size_t i = (size_t)blockIdx.x * 256 + threadIdx.x;
    if (i < n) out[i] = (_Float16)in[i];
}

// ---------------- transpose W[N,K] f32 -> Wt[K,N] f16 ----------------
__global__ __launch_bounds__(256)
void gdn_transpose16(const float* __restrict__ W, _Float16* __restrict__ Wt, int N, int K) {
    int n = blockIdx.x * 32 + (threadIdx.x & 31);
    int k = blockIdx.y * 8 + (threadIdx.x >> 5);
    if (n < N && k < K) Wt[(size_t)k * N + n] = (_Float16)W[(size_t)n * K + k];
}

// ---------------- WMMA GEMM: Y[M,N] f32 = X[M,K] f16 * Wt[K,N] f16 ----------------
// Block: 256 threads = 8 waves; block tile 128(M) x 128(N).
// Wave w owns rows [m0+16w, +16) and all 8 N-tiles: 8 WMMAs per A-fragment load.
// B tile (32 x 128 f16, 8 KB) staged in LDS, shared by all 8 waves.
__global__ __launch_bounds__(256)
void gdn_gemm_wmma(const _Float16* __restrict__ X, const _Float16* __restrict__ Wt,
                   float* __restrict__ Y, int M, int N, int K) {
    __shared__ _Float16 Bs[32 * 128];
    const int wave = threadIdx.x >> 5;
    const int lane = threadIdx.x & 31;
    const int m0   = blockIdx.x * 128 + wave * 16;
    const int n0   = blockIdx.y * 128;
    const int mrow = m0 + (lane & 15);
    const int kb   = (lane < 16) ? 0 : 8;    // A fragment K-base per ISA layout
    const int sr   = threadIdx.x >> 3;       // staging row 0..31
    const int sc   = (threadIdx.x & 7) * 16; // staging col start (16 halves/thread)

    v8f acc[8] = {};
    for (int k0 = 0; k0 < K; k0 += 32) {
        // cooperative B tile stage: Bs[k][n] = Wt[k0+k][n0+n]
        *(v16h*)&Bs[sr * 128 + sc] = *(const v16h*)(Wt + (size_t)(k0 + sr) * N + n0 + sc);
        // A fragment: lane holds row mrow, K = k0+kb+{0..7} and k0+kb+16+{0..7}
        const _Float16* arow = X + (size_t)mrow * K + k0 + kb;
        v8h lo = *(const v8h*)(arow);
        v8h hi = *(const v8h*)(arow + 16);
        v16h a;
#pragma unroll
        for (int i = 0; i < 8; ++i) { a[i] = lo[i]; a[i + 8] = hi[i]; }
        __syncthreads();
#pragma unroll
        for (int t = 0; t < 8; ++t) {
            // B fragment: lane = N column, 16 halves = K (kb+{0..7}, kb+16+{0..7})
            v16h b;
#pragma unroll
            for (int i = 0; i < 8; ++i) {
                b[i]     = Bs[(kb + i) * 128 + t * 16 + (lane & 15)];
                b[i + 8] = Bs[(kb + 16 + i) * 128 + t * 16 + (lane & 15)];
            }
            acc[t] = __builtin_amdgcn_wmma_f32_16x16x32_f16(
                false, a, false, b, (short)0, acc[t], false, false);
        }
        __syncthreads();
    }
    // C/D layout: lane = N (mod 16), VGPR r = M row (+8 for lanes 16..31)
    const int nl = n0 + (lane & 15);
    const int mb = m0 + ((lane < 16) ? 0 : 8);
#pragma unroll
    for (int t = 0; t < 8; ++t)
#pragma unroll
        for (int r = 0; r < 8; ++r)
            Y[(size_t)(mb + r) * N + nl + t * 16] = acc[t][r];
}

// ---------------- g = -exp(A_log)*softplus(x@Wa^T + dt_bias), beta = sigmoid(x@Wb^T) ----------------
__global__ __launch_bounds__(256)
void gdn_gbeta(const float* __restrict__ x, const float* __restrict__ Wa,
               const float* __restrict__ Wb, const float* __restrict__ A_log,
               const float* __restrict__ dt_bias, float* __restrict__ g,
               float* __restrict__ beta) {
    __shared__ float ra[8], rb[8];
    const int m = blockIdx.x;
    const float* xr = x + (size_t)m * HIDD;
    const int wave = threadIdx.x >> 5, lane = threadIdx.x & 31;
    for (int h = 0; h < HH; ++h) {
        float sa = 0.f, sb = 0.f;
        for (int k = threadIdx.x; k < HIDD; k += 256) {
            float xv = xr[k];
            sa += xv * Wa[h * HIDD + k];
            sb += xv * Wb[h * HIDD + k];
        }
#pragma unroll
        for (int off = 16; off > 0; off >>= 1) {
            sa += __shfl_down(sa, off, 32);
            sb += __shfl_down(sb, off, 32);
        }
        if (lane == 0) { ra[wave] = sa; rb[wave] = sb; }
        __syncthreads();
        if (threadIdx.x == 0) {
            float ta = 0.f, tb = 0.f;
#pragma unroll
            for (int i = 0; i < 8; ++i) { ta += ra[i]; tb += rb[i]; }
            float z  = ta + dt_bias[h];
            float sp = (z > 20.f) ? z : log1pf(expf(z));
            g[(size_t)m * HH + h]    = -expf(A_log[h]) * sp;
            beta[(size_t)m * HH + h] = 1.f / (1.f + expf(-tb));
        }
        __syncthreads();
    }
}

// ---------------- depthwise causal conv (K=4) + SiLU ----------------
__global__ __launch_bounds__(256)
void gdn_conv_silu(const float* __restrict__ xp, const float* __restrict__ w,
                   float* __restrict__ y, int C) {
    size_t idx = (size_t)blockIdx.x * 256 + threadIdx.x;
    size_t total = (size_t)MM * C;
    if (idx >= total) return;
    int    c  = (int)(idx % C);
    size_t bt = idx / C;
    int    t  = (int)(bt % TT);
    float acc = 0.f;
#pragma unroll
    for (int j = 0; j < 4; ++j) {
        int tt = t - 3 + j;
        if (tt >= 0) acc += w[c * 4 + j] * xp[(bt + (size_t)(tt - t)) * C + c];
    }
    y[idx] = acc / (1.f + expf(-acc));  // silu
}

// ---------------- l2norm over contiguous 256-float rows (q also scaled by DK^-0.5) ----------------
__global__ __launch_bounds__(256)
void gdn_l2norm(float* __restrict__ buf, float scale) {
    __shared__ float red[8];
    __shared__ float stot;
    size_t base = (size_t)blockIdx.x * DKk;
    float v  = buf[base + threadIdx.x];
    float ss = v * v;
#pragma unroll
    for (int off = 16; off > 0; off >>= 1) ss += __shfl_down(ss, off, 32);
    if ((threadIdx.x & 31) == 0) red[threadIdx.x >> 5] = ss;
    __syncthreads();
    if (threadIdx.x == 0) {
        float t = 0.f;
#pragma unroll
        for (int i = 0; i < 8; ++i) t += red[i];
        stot = t;
    }
    __syncthreads();
    buf[base + threadIdx.x] = v * rsqrtf(stot + 1e-6f) * scale;
}

// ---------------- gated delta-rule recurrence (wave-autonomous, no barriers) ----------------
// Grid: B*H*16 blocks; block owns a DV slice of 32 columns; each wave owns 4 columns.
// Lane layout within wave: kg = lane>>2 (8 groups of 32 K-rows), vsub = lane&3.
// Lane holds S[kg*32 .. kg*32+31][v] in 32 registers. Cross-kg sums via 3x shfl_xor.
__global__ __launch_bounds__(256)
void gdn_deltarule(const float* __restrict__ qb, const float* __restrict__ kbf,
                   const float* __restrict__ vb, const float* __restrict__ g,
                   const float* __restrict__ beta, float* __restrict__ o) {
    const int vblk = blockIdx.x & 15;
    const int bh   = blockIdx.x >> 4;
    const int h    = bh % HH;
    const int b    = bh / HH;
    const int wave = threadIdx.x >> 5;
    const int lane = threadIdx.x & 31;
    const int kg   = lane >> 2;              // 0..7: K group of 32
    const int v    = vblk * 32 + wave * 4 + (lane & 3);  // this lane's DV column

    const float* krow = kbf + (size_t)b * TT * CQK + h * DKk + kg * 32;
    const float* qrow = qb  + (size_t)b * TT * CQK + h * DKk + kg * 32;
    const float* vrow = vb  + (size_t)b * TT * CV  + h * DVv + v;
    const float* grow = g    + (size_t)b * TT * HH + h;
    const float* brow = beta + (size_t)b * TT * HH + h;
    float*       orow = o   + (size_t)b * TT * CV  + h * DVv + v;

    float S[32];
#pragma unroll
    for (int i = 0; i < 32; ++i) S[i] = 0.f;

    for (int t = 0; t < TT; ++t) {
        // load this lane's K/Q slices (cacheline-broadcast across the 4 vsub lanes)
        float kreg[32], qreg[32];
#pragma unroll
        for (int j = 0; j < 8; ++j) {
            v4f kv = *(const v4f*)(krow + j * 4);
            v4f qv = *(const v4f*)(qrow + j * 4);
#pragma unroll
            for (int e = 0; e < 4; ++e) { kreg[j * 4 + e] = kv[e]; qreg[j * 4 + e] = qv[e]; }
        }
        const float vval = *vrow;
        const float dec  = expf(*grow);
        const float bt   = *brow;
        // prefetch next timestep's rows (addresses fully predictable)
        __builtin_prefetch(krow + CQK, 0, 0);
        __builtin_prefetch(qrow + CQK, 0, 0);

        // decay + partial k.S
        float part = 0.f;
#pragma unroll
        for (int i = 0; i < 32; ++i) {
            S[i] *= dec;
            part += kreg[i] * S[i];
        }
        // reduce over kg (lane bits 2..4)
        part += __shfl_xor(part, 4, 32);
        part += __shfl_xor(part, 8, 32);
        part += __shfl_xor(part, 16, 32);
        const float dl = (vval - part) * bt;

        // rank-1 update + partial q.S
        float op = 0.f;
#pragma unroll
        for (int i = 0; i < 32; ++i) {
            S[i] += kreg[i] * dl;
            op   += qreg[i] * S[i];
        }
        op += __shfl_xor(op, 4, 32);
        op += __shfl_xor(op, 8, 32);
        op += __shfl_xor(op, 16, 32);
        if (kg == 0) *orow = op;

        krow += CQK; qrow += CQK; vrow += CV; orow += CV;
        grow += HH;  brow += HH;
    }
}

// ---------------- og = o * silu(gate); RMS-norm over DV; * norm_w; -> f16 ----------------
__global__ __launch_bounds__(256)
void gdn_gatenorm(const float* __restrict__ o, const float* __restrict__ gatep,
                  const float* __restrict__ nw, _Float16* __restrict__ og) {
    __shared__ float red[8];
    __shared__ float sscale;
    const size_t base = (size_t)blockIdx.x * DVv;  // (m*H+h)*512 == row-major [M, H*DV]
    float val[2];
    float ss = 0.f;
#pragma unroll
    for (int i = 0; i < 2; ++i) {
        int   v  = threadIdx.x + i * 256;
        float gv = gatep[base + v];
        float ov = o[base + v] * (gv / (1.f + expf(-gv)));
        val[i] = ov;
        ss += ov * ov;
    }
#pragma unroll
    for (int off = 16; off > 0; off >>= 1) ss += __shfl_down(ss, off, 32);
    if ((threadIdx.x & 31) == 0) red[threadIdx.x >> 5] = ss;
    __syncthreads();
    if (threadIdx.x == 0) {
        float t = 0.f;
#pragma unroll
        for (int i = 0; i < 8; ++i) t += red[i];
        sscale = rsqrtf(t / (float)DVv + 1e-5f);
    }
    __syncthreads();
#pragma unroll
    for (int i = 0; i < 2; ++i) {
        int v = threadIdx.x + i * 256;
        og[base + v] = (_Float16)(val[i] * sscale * nw[v]);
    }
}

extern "C" void kernel_launch(void* const* d_in, const int* in_sizes, int n_in,
                              void* d_out, int out_size, void* d_ws, size_t ws_size,
                              hipStream_t stream) {
    (void)in_sizes; (void)n_in; (void)out_size; (void)ws_size;
    const float* x        = (const float*)d_in[0];
    const float* Wq       = (const float*)d_in[1];
    const float* Wk       = (const float*)d_in[2];
    const float* Wv       = (const float*)d_in[3];
    const float* Wa       = (const float*)d_in[4];
    const float* Wb       = (const float*)d_in[5];
    const float* A_log    = (const float*)d_in[6];
    const float* dt_bias  = (const float*)d_in[7];
    const float* conv_q_w = (const float*)d_in[8];
    const float* conv_k_w = (const float*)d_in[9];
    const float* conv_v_w = (const float*)d_in[10];
    const float* Wg       = (const float*)d_in[11];
    const float* norm_w   = (const float*)d_in[12];
    const float* Wo       = (const float*)d_in[13];
    float* out = (float*)d_out;

    // bump allocator over workspace (256B aligned)
    uint8_t* p = (uint8_t*)d_ws;
    auto take = [&](size_t bytes) -> void* {
        void* r = (void*)p;
        p += (bytes + 255) & ~(size_t)255;
        return r;
    };
    _Float16* x16  = (_Float16*)take((size_t)MM * HIDD * 2);
    _Float16* WqT  = (_Float16*)take((size_t)HIDD * CQK * 2);
    _Float16* WkT  = (_Float16*)take((size_t)HIDD * CQK * 2);
    _Float16* WvT  = (_Float16*)take((size_t)HIDD * CV * 2);
    _Float16* WgT  = (_Float16*)take((size_t)HIDD * CV * 2);
    _Float16* WoT  = (_Float16*)take((size_t)CV * HIDD * 2);
    float* qpre = (float*)take((size_t)MM * CQK * 4);
    float* kpre = (float*)take((size_t)MM * CQK * 4);
    float* vpre = (float*)take((size_t)MM * CV * 4);
    float* gpre = (float*)take((size_t)MM * CV * 4);
    float* qbuf = (float*)take((size_t)MM * CQK * 4);
    float* kbuf = (float*)take((size_t)MM * CQK * 4);
    float* vbuf = (float*)take((size_t)MM * CV * 4);
    float* gdec = (float*)take((size_t)MM * HH * 4);
    float* bet  = (float*)take((size_t)MM * HH * 4);
    float*    obuf = vpre;              // vpre dead after conv_v
    _Float16* og16 = (_Float16*)qpre;   // qpre dead after conv_q (same byte size)

    // 1. cast activations, transpose+cast weights to f16 [K,N]
    size_t nx = (size_t)MM * HIDD;
    gdn_cast16<<<(unsigned)((nx + 255) / 256), 256, 0, stream>>>(x, x16, nx);
    gdn_transpose16<<<dim3(CQK / 32, HIDD / 8), 256, 0, stream>>>(Wq, WqT, CQK, HIDD);
    gdn_transpose16<<<dim3(CQK / 32, HIDD / 8), 256, 0, stream>>>(Wk, WkT, CQK, HIDD);
    gdn_transpose16<<<dim3(CV / 32, HIDD / 8), 256, 0, stream>>>(Wv, WvT, CV, HIDD);
    gdn_transpose16<<<dim3(CV / 32, HIDD / 8), 256, 0, stream>>>(Wg, WgT, CV, HIDD);
    gdn_transpose16<<<dim3(HIDD / 32, CV / 8), 256, 0, stream>>>(Wo, WoT, HIDD, CV);

    // 2. WMMA projection GEMMs (block tile 128x128)
    gdn_gemm_wmma<<<dim3(MM / 128, CQK / 128), 256, 0, stream>>>(x16, WqT, qpre, MM, CQK, HIDD);
    gdn_gemm_wmma<<<dim3(MM / 128, CQK / 128), 256, 0, stream>>>(x16, WkT, kpre, MM, CQK, HIDD);
    gdn_gemm_wmma<<<dim3(MM / 128, CV / 128), 256, 0, stream>>>(x16, WvT, vpre, MM, CV, HIDD);
    gdn_gemm_wmma<<<dim3(MM / 128, CV / 128), 256, 0, stream>>>(x16, WgT, gpre, MM, CV, HIDD);

    // 3. per-head decay / beta
    gdn_gbeta<<<MM, 256, 0, stream>>>(x, Wa, Wb, A_log, dt_bias, gdec, bet);

    // 4. causal depthwise conv + silu, then L2-norm q (with DK^-0.5) and k
    gdn_conv_silu<<<(unsigned)(((size_t)MM * CQK + 255) / 256), 256, 0, stream>>>(qpre, conv_q_w, qbuf, CQK);
    gdn_conv_silu<<<(unsigned)(((size_t)MM * CQK + 255) / 256), 256, 0, stream>>>(kpre, conv_k_w, kbuf, CQK);
    gdn_conv_silu<<<(unsigned)(((size_t)MM * CV + 255) / 256), 256, 0, stream>>>(vpre, conv_v_w, vbuf, CV);
    gdn_l2norm<<<MM * HH, 256, 0, stream>>>(qbuf, 0.0625f);  // DK^-0.5 = 1/16
    gdn_l2norm<<<MM * HH, 256, 0, stream>>>(kbuf, 1.0f);

    // 5. sequential gated delta-rule (192 independent blocks, barrier-free inner loop)
    gdn_deltarule<<<BB * HH * 16, 256, 0, stream>>>(qbuf, kbuf, vbuf, gdec, bet, obuf);

    // 6. gate + RMS-norm -> f16
    gdn_gatenorm<<<MM * HH, 256, 0, stream>>>(obuf, gpre, norm_w, og16);

    // 7. output WMMA GEMM -> d_out
    gdn_gemm_wmma<<<dim3(MM / 128, HIDD / 128), 256, 0, stream>>>(og16, WoT, out, MM, HIDD, CV);
}
